// MultiHeadSelfAttention_33715493273791
// MI455X (gfx1250) — matrix operand order repaired and checked by reading, not compiled, not run on hardware
//
#include <hip/hip_runtime.h>
#include <hip/hip_bf16.h>

// ---------------------------------------------------------------------------
// MultiHeadSelfAttention for gfx1250 (MI455X): bf16 WMMA everywhere.
//   stage 1: qkv(bf16)  = x(f32) @ w_qkv(f32)          [4096 x 2304]
//   stage 2: ctx(bf16)  = flash-attention(qkv)         [4096 x 768]
//   stage 3: out(f32)   = ctx(bf16) @ w_proj(f32)      [4096 x 768]
// Round-2 changes: DPP16 butterfly reductions (no ds_bpermute / dscnt
// drains), 1/sqrt(hd) folded into Q fragments, vectorized b128 staging with
// register double-buffering to overlap HBM with the WMMA pipe.
// ---------------------------------------------------------------------------

typedef __bf16 bf16;
typedef __attribute__((ext_vector_type(16))) __bf16 v16bf;
typedef __attribute__((ext_vector_type(8)))  __bf16 v8bf;
typedef __attribute__((ext_vector_type(4)))  __bf16 v4bf;
typedef __attribute__((ext_vector_type(8)))  float  v8f;
typedef __attribute__((ext_vector_type(4)))  float  v4f;

#define EMBED  768
#define NHEADS 12
#define HD     64
#define SEQ    2048
#define BATCH  2
#define QKVC   (3 * EMBED) /* 2304 */

template <typename T> struct vec4;
template <> struct vec4<float> { typedef v4f type; };
template <> struct vec4<bf16>  { typedef v4bf type; };

// ---- WMMA wrapper: D = A(16x32 bf16) * B(32x16 bf16) + C(16x16 f32) -------
static __device__ inline v8f wmma_bf16(v16bf a, v16bf b, v8f c) {
  return __builtin_amdgcn_wmma_f32_16x16x32_bf16(
      /*neg_a=*/false, a, /*neg_b=*/false, b,
      /*c_mod=*/(short)0, c, /*reuse_a=*/false, /*reuse_b=*/false);
}

// A-operand fragment (16x32, row-major source, 16-bit elements).
static __device__ inline v16bf load_a_frag(const bf16* base, int ld, int lane,
                                           int koff) {
  const int row = lane & 15;
  const int kb  = koff + ((lane >> 4) << 3);
  const bf16* p = base + row * ld + kb;
  v8bf lo = *(const v8bf*)(p);
  v8bf hi = *(const v8bf*)(p + 16);
  return __builtin_shufflevector(lo, hi, 0, 1, 2, 3, 4, 5, 6, 7,
                                 8, 9, 10, 11, 12, 13, 14, 15);
}

// B-operand fragment (32x16) read from a K-contiguous (B^T, [n][k]) buffer.
static __device__ inline v16bf load_bt_frag(const bf16* base, int ld, int lane,
                                            int koff) {
  const int col = lane & 15;
  const int kb  = koff + ((lane >> 4) << 3);
  const bf16* p = base + col * ld + kb;
  v8bf lo = *(const v8bf*)(p);
  v8bf hi = *(const v8bf*)(p + 16);
  return __builtin_shufflevector(lo, hi, 0, 1, 2, 3, 4, 5, 6, 7,
                                 8, 9, 10, 11, 12, 13, 14, 15);
}

// ---- DPP16 butterfly over 16-lane groups: pure VALU, no LDS, no waits -----
// xor1 = quad_perm(1,0,3,2)=0xB1, xor2 = quad_perm(2,3,0,1)=0x4E,
// then row_half_mirror (0x141) merges quads, row_mirror (0x140) merges halves.
template <int CTRL>
static __device__ inline float dpp_mov(float v) {
  return __builtin_bit_cast(
      float, __builtin_amdgcn_update_dpp(0, __builtin_bit_cast(int, v), CTRL,
                                         0xF, 0xF, true));
}
static __device__ inline float redmax16(float v) {
  v = fmaxf(v, dpp_mov<0xB1>(v));
  v = fmaxf(v, dpp_mov<0x4E>(v));
  v = fmaxf(v, dpp_mov<0x141>(v));
  v = fmaxf(v, dpp_mov<0x140>(v));
  return v;
}
static __device__ inline float redsum16(float v) {
  v += dpp_mov<0xB1>(v);
  v += dpp_mov<0x4E>(v);
  v += dpp_mov<0x141>(v);
  v += dpp_mov<0x140>(v);
  return v;
}

// ---------------------------------------------------------------------------
// Generic tiled GEMM: C[M,N] = A[M,K] * B[K,N].  Block tile 128x128, K-step
// 32, 8 waves (4x2), each wave computes 32x64 (2x4 WMMA fragments).
// Register double-buffered staging: global b128 loads for tile kt+1 are in
// flight while tile kt runs through the WMMA pipe.
// ---------------------------------------------------------------------------
template <typename AT, typename CT>
__global__ __launch_bounds__(256) void gemm_wmma(const AT* __restrict__ A,
                                                 const float* __restrict__ B,
                                                 CT* __restrict__ C,
                                                 int M, int N, int K) {
  __shared__ alignas(16) bf16 sA[128][40];   // [m][k], padded
  __shared__ alignas(16) bf16 sBt[128][40];  // [n][k], padded (B transposed)

  const int tid  = threadIdx.x;
  const int wave = tid >> 5;
  const int lane = tid & 31;
  const int wm   = wave >> 1;  // 0..3 -> 32-row strip
  const int wn   = wave & 1;   // 0..1 -> 64-col strip

  const int row0 = blockIdx.y * 128;
  const int col0 = blockIdx.x * 128;

  typename vec4<AT>::type aR[4];
  v4f bR[4];

  auto load_tiles = [&](int kt) {
#pragma unroll
    for (int it = 0; it < 4; ++it) {
      const int i = tid * 4 + it * 1024;
      const int m = i >> 5, k = i & 31;          // A: 128x32
      aR[it] = *(const typename vec4<AT>::type*)&A[(size_t)(row0 + m) * K +
                                                   (kt << 5) + k];
      const int kk = i >> 7, n = i & 127;        // B: 32x128
      bR[it] = *(const v4f*)&B[(size_t)((kt << 5) + kk) * N + col0 + n];
    }
  };
  auto store_tiles = [&]() {
#pragma unroll
    for (int it = 0; it < 4; ++it) {
      const int i = tid * 4 + it * 1024;
      const int m = i >> 5, k = i & 31;
      v4bf av;
      av[0] = (bf16)aR[it][0]; av[1] = (bf16)aR[it][1];
      av[2] = (bf16)aR[it][2]; av[3] = (bf16)aR[it][3];
      *(v4bf*)&sA[m][k] = av;
      const int kk = i >> 7, n = i & 127;
      sBt[n + 0][kk] = (bf16)bR[it][0];
      sBt[n + 1][kk] = (bf16)bR[it][1];
      sBt[n + 2][kk] = (bf16)bR[it][2];
      sBt[n + 3][kk] = (bf16)bR[it][3];
    }
  };

  v8f acc[2][4];
#pragma unroll
  for (int r = 0; r < 2; ++r)
#pragma unroll
    for (int c = 0; c < 4; ++c) acc[r][c] = (v8f)0.0f;

  const int ksteps = K >> 5;
  load_tiles(0);
  for (int kt = 0; kt < ksteps; ++kt) {
    store_tiles();
    __syncthreads();
    if (kt + 1 < ksteps) load_tiles(kt + 1);  // overlap HBM with WMMA

    v16bf aF[2], bF[4];
#pragma unroll
    for (int r = 0; r < 2; ++r)
      aF[r] = load_a_frag(&sA[wm * 32 + r * 16][0], 40, lane, 0);
#pragma unroll
    for (int c = 0; c < 4; ++c)
      bF[c] = load_bt_frag(&sBt[wn * 64 + c * 16][0], 40, lane, 0);

#pragma unroll
    for (int r = 0; r < 2; ++r)
#pragma unroll
      for (int c = 0; c < 4; ++c) acc[r][c] = wmma_bf16(aF[r], bF[c], acc[r][c]);

    __syncthreads();
  }

  // Epilogue: slot s holds M = s + 8*(lane/16), N = lane%16.
  const int half = lane >> 4, coln = lane & 15;
#pragma unroll
  for (int r = 0; r < 2; ++r)
#pragma unroll
    for (int c = 0; c < 4; ++c) {
      const int gm0 = row0 + wm * 32 + r * 16 + 8 * half;
      const int gn  = col0 + wn * 64 + c * 16 + coln;
#pragma unroll
      for (int s = 0; s < 8; ++s)
        C[(size_t)(gm0 + s) * N + gn] = (CT)acc[r][c][s];
    }
}

// ---------------------------------------------------------------------------
// Fused flash attention.  Grid: (SEQ/128, NHEADS, BATCH), 256 threads.
// Each wave owns 16 query rows x HD=64; key blocks of 64 streamed via LDS
// with register double-buffering.  1/sqrt(hd) is pre-folded into Q.
// ---------------------------------------------------------------------------
__global__ __launch_bounds__(256) void attn_wmma(const bf16* __restrict__ qkv,
                                                 bf16* __restrict__ ctx) {
  __shared__ alignas(16) bf16 sK[64][72];     // K rows  == B^T for Q*K^T
  __shared__ alignas(16) bf16 sVt[64][72];    // V^T     == B^T for P*V
  __shared__ alignas(16) bf16 sP[8][16][72];  // per-wave P round-trip

  const int tid  = threadIdx.x;
  const int wave = tid >> 5;
  const int lane = tid & 31;
  const int half = lane >> 4, coln = lane & 15;

  const int h = blockIdx.y;
  const int b = blockIdx.z;
  const int qrow0 = blockIdx.x * 128 + wave * 16;
  const size_t rowbase = (size_t)(b * SEQ) * QKVC;

  // Q fragments with the softmax scale folded in (exact: *2^-3 in bf16).
  v16bf aQ[2];
#pragma unroll
  for (int kc = 0; kc < 2; ++kc) {
    v16bf q = load_a_frag(qkv + rowbase + (size_t)qrow0 * QKVC + h * HD, QKVC,
                          lane, kc * 32);
    aQ[kc] = q * (bf16)0.125f;  // v_pk_mul_bf16
  }

  v4bf kR[4], vR[4];
  auto load_kv = [&](int kb0) {
#pragma unroll
    for (int it = 0; it < 4; ++it) {
      const int i = tid * 4 + it * 1024;
      const int n = i >> 6, d = i & 63;
      const size_t r = rowbase + (size_t)(kb0 + n) * QKVC + h * HD + d;
      kR[it] = *(const v4bf*)(qkv + r + EMBED);
      vR[it] = *(const v4bf*)(qkv + r + 2 * EMBED);
    }
  };
  auto store_kv = [&]() {
#pragma unroll
    for (int it = 0; it < 4; ++it) {
      const int i = tid * 4 + it * 1024;
      const int n = i >> 6, d = i & 63;
      *(v4bf*)&sK[n][d] = kR[it];
      sVt[d + 0][n] = vR[it][0];
      sVt[d + 1][n] = vR[it][1];
      sVt[d + 2][n] = vR[it][2];
      sVt[d + 3][n] = vR[it][3];
    }
  };

  float mrow[8], lrow[8];
  v8f accO[4];
#pragma unroll
  for (int s = 0; s < 8; ++s) { mrow[s] = -1e30f; lrow[s] = 0.0f; }
#pragma unroll
  for (int d = 0; d < 4; ++d) accO[d] = (v8f)0.0f;

  load_kv(0);
  for (int kb0 = 0; kb0 < SEQ; kb0 += 64) {
    store_kv();
    __syncthreads();
    if (kb0 + 64 < SEQ) load_kv(kb0 + 64);  // overlap HBM with compute

    // S = (Q/sqrt(hd)) * K^T  (16 x 64 per wave).
    v8f sfr[4];
#pragma unroll
    for (int c = 0; c < 4; ++c) {
      v8f s = (v8f)0.0f;
#pragma unroll
      for (int kc = 0; kc < 2; ++kc) {
        v16bf bk = load_bt_frag(&sK[c * 16][0], 72, lane, kc * 32);
        s = wmma_bf16(aQ[kc], bk, s);
      }
      sfr[c] = s;
    }

    // Online softmax; all reductions are DPP16 VALU butterflies.
#pragma unroll
    for (int s = 0; s < 8; ++s) {
      float tm = fmaxf(fmaxf(sfr[0][s], sfr[1][s]),
                       fmaxf(sfr[2][s], sfr[3][s]));
      tm = redmax16(tm);
      const float nm   = fmaxf(mrow[s], tm);
      const float corr = __expf(mrow[s] - nm);
      float ps = 0.0f;
#pragma unroll
      for (int c = 0; c < 4; ++c) {
        const float p = __expf(sfr[c][s] - nm);
        sfr[c][s] = p;
        ps += p;
      }
      ps = redsum16(ps);
      lrow[s] = lrow[s] * corr + ps;
      mrow[s] = nm;
#pragma unroll
      for (int d = 0; d < 4; ++d) accO[d][s] *= corr;
    }

    // P: C-fragment layout -> per-wave LDS patch -> A-fragment layout.
#pragma unroll
    for (int c = 0; c < 4; ++c)
#pragma unroll
      for (int s = 0; s < 8; ++s)
        sP[wave][s + 8 * half][c * 16 + coln] = (bf16)sfr[c][s];
    asm volatile("s_wait_dscnt 0" ::: "memory");  // wave-private ordering

    // O += P * V
#pragma unroll
    for (int kc = 0; kc < 2; ++kc) {
      v16bf pA = load_a_frag(&sP[wave][0][0], 72, lane, kc * 32);
#pragma unroll
      for (int d = 0; d < 4; ++d) {
        v16bf bv = load_bt_frag(&sVt[d * 16][0], 72, lane, kc * 32);
        accO[d] = wmma_bf16(pA, bv, accO[d]);
      }
    }
    __syncthreads();
  }

  // Normalize and write ctx[b*SEQ+q][h*HD+d] as bf16.
#pragma unroll
  for (int s = 0; s < 8; ++s) {
    const float invl = 1.0f / lrow[s];
    const int qr = qrow0 + s + 8 * half;
#pragma unroll
    for (int d = 0; d < 4; ++d) {
      const int dc = d * 16 + coln;
      ctx[(size_t)(b * SEQ + qr) * EMBED + h * HD + dc] =
          (bf16)(accO[d][s] * invl);
    }
  }
}

// ---------------------------------------------------------------------------
extern "C" void kernel_launch(void* const* d_in, const int* in_sizes, int n_in,
                              void* d_out, int out_size, void* d_ws,
                              size_t ws_size, hipStream_t stream) {
  const float* x      = (const float*)d_in[0];
  const float* w_qkv  = (const float*)d_in[1];
  const float* w_proj = (const float*)d_in[2];
  float* out = (float*)d_out;

  bf16* qkv = (bf16*)d_ws;
  bf16* ctx = (bf16*)((char*)d_ws + (size_t)(BATCH * SEQ) * QKVC * sizeof(bf16));

  const int M = BATCH * SEQ;  // 4096
  dim3 blk(256);

  // 1) qkv = x @ w_qkv   (f32 in, bf16 out)
  gemm_wmma<float, bf16>
      <<<dim3(QKVC / 128, M / 128), blk, 0, stream>>>(x, w_qkv, qkv, M, QKVC,
                                                      EMBED);
  // 2) fused flash attention -> ctx (bf16)
  attn_wmma<<<dim3(SEQ / 128, NHEADS, BATCH), blk, 0, stream>>>(qkv, ctx);

  // 3) out = ctx @ w_proj  (bf16 A, f32 out)
  gemm_wmma<bf16, float>
      <<<dim3(EMBED / 128, M / 128), blk, 0, stream>>>(ctx, w_proj, out, M,
                                                       EMBED, EMBED);
}
